// PerformerAttention_64656437674194
// MI455X (gfx1250) — matrix operand order repaired and checked by reading, not compiled, not run on hardware
//
#include <hip/hip_runtime.h>

typedef __attribute__((ext_vector_type(16))) __bf16 v16bf;
typedef __attribute__((ext_vector_type(8)))  __bf16 v8bf;
typedef __attribute__((ext_vector_type(8)))  float  v8f;
typedef __attribute__((ext_vector_type(4)))  int    v4i;

#define AS1 __attribute__((address_space(1)))
#define AS3 __attribute__((address_space(3)))

#if defined(__has_builtin)
#if __has_builtin(__builtin_amdgcn_global_load_async_to_lds_b128) && \
    __has_builtin(__builtin_amdgcn_s_wait_asynccnt)
#define USE_ASYNC_LDS 1
#endif
#endif
#ifndef USE_ASYNC_LDS
#define USE_ASYNC_LDS 0
#endif

// Problem constants
constexpr int NSEQ  = 4096;
constexpr int CDIM  = 1024;
constexpr int HEADS = 16;
constexpr int HC    = 64;
constexpr int TOK   = 4 * NSEQ;        // 16384 tokens (B*N)
constexpr int BH    = 4 * HEADS;       // 64 (batch*heads)
constexpr int RP    = 320;             // R=266 padded to multiple of 64
constexpr int RREAL = 266;
constexpr float FEPS = 1e-3f;
constexpr int KCHUNKS = 8;
constexpr int KCLEN   = NSEQ / KCHUNKS; // 512

static __device__ __forceinline__ __bf16 tob(float f) { return (__bf16)f; }

struct EpiArgs {
  void* out0;
  void* out1;
  const float* fin0;
};

struct GemmArgs {
  const __bf16* A;  int lda; long sAb, sAh;   // A row-major M x K (+ per-b / per-h strides)
  const __bf16* Bt; int ldb; long sBb, sBh;   // B^T row-major N x K
  int K;        // K length of one chunk
  int kChunks;  // split-K chunks per bh
  EpiArgs ep;
};

// ---------------- epilogues (per-element) ----------------
struct EpiQKV { // n<2048 -> QK buffer (TOK x 2048); n>=2048 -> vT[(b*16+h)*64+d][nseq]
  static __device__ __forceinline__ void op(const EpiArgs& e, int, int, int m, int n, float v) {
    __bf16 bv = tob(v);
    if (n < 2048) {
      ((__bf16*)e.out0)[(long)m * 2048 + n] = bv;
    } else {
      int j = n - 2048, h = j >> 6, d = j & 63, b = m >> 12, ns = m & 4095;
      ((__bf16*)e.out1)[(((long)(b * 16 + h)) * 64 + d) * 4096 + ns] = bv;
    }
  }
};
struct EpiQP { // relu+eps, zero padded cols; qp[bh][m][r] row-major
  static __device__ __forceinline__ void op(const EpiArgs& e, int, int bh, int m, int n, float v) {
    float r = (n < RREAL) ? (fmaxf(v, 0.0f) + FEPS) : 0.0f;
    ((__bf16*)e.out0)[((long)bh * NSEQ + m) * RP + n] = tob(r);
  }
};
struct EpiKP { // transposed store: kpT[bh][r][m]
  static __device__ __forceinline__ void op(const EpiArgs& e, int, int bh, int m, int n, float v) {
    float r = (n < RREAL) ? (fmaxf(v, 0.0f) + FEPS) : 0.0f;
    ((__bf16*)e.out0)[(long)bh * RP * NSEQ + (long)n * NSEQ + m] = tob(r);
  }
};
struct EpiKV { // split-K partials: part[z][m][n] fp32 (z = bh*KCHUNKS + c)
  static __device__ __forceinline__ void op(const EpiArgs& e, int z, int, int m, int n, float v) {
    ((float*)e.out0)[(long)z * (HC * RP) + (long)m * RP + n] = v;
  }
};
struct EpiCtx { // ctx[b*4096+m][h*64+n] = v * dinv[bh][m]
  static __device__ __forceinline__ void op(const EpiArgs& e, int, int bh, int m, int n, float v) {
    int b = bh >> 4, h = bh & 15;
    float dv = e.fin0[(long)bh * NSEQ + m];
    ((__bf16*)e.out0)[((long)(b * NSEQ + m)) * CDIM + h * 64 + n] = tob(v * dv);
  }
};
struct EpiOut { // fp32 + bias
  static __device__ __forceinline__ void op(const EpiArgs& e, int, int, int m, int n, float v) {
    ((float*)e.out0)[(long)m * CDIM + n] = v + e.fin0[n];
  }
};

// ---------------- templated bf16 WMMA GEMM ----------------
template <int BM, int BN, int WM, int WN, class Epi>
__launch_bounds__((BM / WM) * (BN / WN) * 32)
__global__ void gemm_bf16_wmma(GemmArgs g) {
  constexpr int BK   = 64;
  constexpr int NWV  = (BM / WM) * (BN / WN);
  constexpr int NTHR = NWV * 32;
  constexpr int MT   = WM / 16;
  constexpr int NT2  = WN / 16;
  constexpr int LK   = BK + 8;           // +16B pad to spread LDS banks
  constexpr int SROWS = NTHR / 8;        // tile rows staged per unrolled step
  constexpr int AITER = BM / SROWS;
  constexpr int BITER = BN / SROWS;
  constexpr int NLD   = AITER + BITER;   // async loads per tile per thread
  static_assert(AITER * SROWS == BM, "A staging must divide");
  static_assert(BITER * SROWS == BN, "B staging must divide");

#if USE_ASYNC_LDS
  __shared__ __attribute__((aligned(16))) __bf16 sA[2][BM][LK];
  __shared__ __attribute__((aligned(16))) __bf16 sB[2][BN][LK];
#else
  __shared__ __attribute__((aligned(16))) __bf16 sA[1][BM][LK];
  __shared__ __attribute__((aligned(16))) __bf16 sB[1][BN][LK];
#endif

  const int tid  = threadIdx.x;
  const int lane = tid & 31;
  const int w    = tid >> 5;
  const int wm   = w % (BM / WM);
  const int wn   = w / (BM / WM);

  const int z  = blockIdx.z;
  const int c  = z % g.kChunks;
  const int bh = z / g.kChunks;
  const int b  = bh >> 4, h = bh & 15;

  const int m0 = blockIdx.y * BM;
  const int n0 = blockIdx.x * BN;

  // Single per-thread base pointers; per-iteration offsets are uniform or constexpr.
  const int rS = tid >> 3, cS = (tid & 7) << 3;  // staging row / col(8 bf16)
  const __bf16* aCur = g.A + (long)b * g.sAb + (long)h * g.sAh + (long)c * g.K +
                       (long)(m0 + rS) * g.lda + cS;
  const __bf16* bCur = g.Bt + (long)b * g.sBb + (long)h * g.sBh + (long)c * g.K +
                       (long)(n0 + rS) * g.ldb + cS;
  const long aStep = (long)SROWS * g.lda;   // uniform
  const long bStep = (long)SROWS * g.ldb;   // uniform

  v8f acc[MT][NT2];
#pragma unroll
  for (int i = 0; i < MT; ++i)
#pragma unroll
    for (int j = 0; j < NT2; ++j) {
      v8f zv = {0.f, 0.f, 0.f, 0.f, 0.f, 0.f, 0.f, 0.f};
      acc[i][j] = zv;
    }

  const int lrow = lane & 15;
  const int aSel = (lane < 16) ? 0 : 8;   // A: K chunks {aSel..+7, aSel+16..+23}
  const int bSel = (lane < 16) ? 0 : 8;   // B: K chunks {bSel..+7, bSel+16..+23}
  const int mOff = (lane < 16) ? 0 : 8;

#if USE_ASYNC_LDS
  // CDNA5 async global->LDS: no VGPR staging, tracked by ASYNCcnt.
  __bf16* sAB0 = &sA[0][rS][cS];
  __bf16* sBB0 = &sB[0][rS][cS];
  __bf16* sAB1 = &sA[1][rS][cS];
  __bf16* sBB1 = &sB[1][rS][cS];
  auto issueTile = [&](int buf) {
    __bf16* sa = buf ? sAB1 : sAB0;
    __bf16* sb = buf ? sBB1 : sBB0;
#pragma unroll
    for (int it = 0; it < AITER; ++it)
      __builtin_amdgcn_global_load_async_to_lds_b128(
          (AS1 v4i*)(void*)(aCur + (long)it * aStep),
          (AS3 v4i*)(void*)(sa + it * (SROWS * LK)), 0, 0);
#pragma unroll
    for (int it = 0; it < BITER; ++it)
      __builtin_amdgcn_global_load_async_to_lds_b128(
          (AS1 v4i*)(void*)(bCur + (long)it * bStep),
          (AS3 v4i*)(void*)(sb + it * (SROWS * LK)), 0, 0);
    aCur += BK;
    bCur += BK;
  };

  const int T = g.K / BK;
  issueTile(0);
  for (int t = 0; t < T; ++t) {
    if (t + 1 < T) {
      issueTile((t + 1) & 1);
      __builtin_amdgcn_s_wait_asynccnt(NLD);  // current tile's loads complete
    } else {
      __builtin_amdgcn_s_wait_asynccnt(0);
    }
    __syncthreads();  // all waves' current tile visible in LDS

    const __bf16* aF = &sA[t & 1][wm * WM + lrow][aSel];
    const __bf16* bF = &sB[t & 1][wn * WN + lrow][bSel];
#pragma unroll
    for (int kk = 0; kk < BK; kk += 32) {
      v16bf af[MT], bfr[NT2];
#pragma unroll
      for (int tm = 0; tm < MT; ++tm) {
        v8bf lo = *(const v8bf*)(aF + tm * (16 * LK) + kk);
        v8bf hi = *(const v8bf*)(aF + tm * (16 * LK) + kk + 16);
        af[tm] = __builtin_shufflevector(lo, hi, 0, 1, 2, 3, 4, 5, 6, 7, 8, 9, 10, 11, 12, 13, 14, 15);
      }
#pragma unroll
      for (int tn = 0; tn < NT2; ++tn) {
        v8bf lo = *(const v8bf*)(bF + tn * (16 * LK) + kk);
        v8bf hi = *(const v8bf*)(bF + tn * (16 * LK) + kk + 16);
        bfr[tn] = __builtin_shufflevector(lo, hi, 0, 1, 2, 3, 4, 5, 6, 7, 8, 9, 10, 11, 12, 13, 14, 15);
      }
#pragma unroll
      for (int tm = 0; tm < MT; ++tm)
#pragma unroll
        for (int tn = 0; tn < NT2; ++tn)
          acc[tm][tn] = __builtin_amdgcn_wmma_f32_16x16x32_bf16(
              false, af[tm], false, bfr[tn], (short)0, acc[tm][tn], false, false);
    }
    __syncthreads();  // all waves done reading this buffer before it is refilled
  }
#else
  // Fallback: synchronous staging; memory fences stop the pipeliner from
  // extending staged-register live ranges across compute (which caused spills).
  __bf16* sABase = &sA[0][rS][cS];
  __bf16* sBBase = &sB[0][rS][cS];
  const __bf16* aF = &sA[0][wm * WM + lrow][aSel];
  const __bf16* bF = &sB[0][wn * WN + lrow][bSel];

  for (int k0 = 0; k0 < g.K; k0 += BK) {
    asm volatile("" ::: "memory");
    {
      int4 ar[AITER], br[BITER];
#pragma unroll
      for (int it = 0; it < AITER; ++it) ar[it] = *(const int4*)(aCur + (long)it * aStep);
#pragma unroll
      for (int it = 0; it < BITER; ++it) br[it] = *(const int4*)(bCur + (long)it * bStep);
      aCur += BK; bCur += BK;
      if (k0 + BK < g.K) {
        __builtin_prefetch(aCur, 0, 1);
        __builtin_prefetch(bCur, 0, 1);
      }
      __syncthreads();
#pragma unroll
      for (int it = 0; it < AITER; ++it) *(int4*)(sABase + it * (SROWS * LK)) = ar[it];
#pragma unroll
      for (int it = 0; it < BITER; ++it) *(int4*)(sBBase + it * (SROWS * LK)) = br[it];
    }
    asm volatile("" ::: "memory");
    __syncthreads();

#pragma unroll
    for (int kk = 0; kk < BK; kk += 32) {
      v16bf af[MT], bfr[NT2];
#pragma unroll
      for (int tm = 0; tm < MT; ++tm) {
        v8bf lo = *(const v8bf*)(aF + tm * (16 * LK) + kk);
        v8bf hi = *(const v8bf*)(aF + tm * (16 * LK) + kk + 16);
        af[tm] = __builtin_shufflevector(lo, hi, 0, 1, 2, 3, 4, 5, 6, 7, 8, 9, 10, 11, 12, 13, 14, 15);
      }
#pragma unroll
      for (int tn = 0; tn < NT2; ++tn) {
        v8bf lo = *(const v8bf*)(bF + tn * (16 * LK) + kk);
        v8bf hi = *(const v8bf*)(bF + tn * (16 * LK) + kk + 16);
        bfr[tn] = __builtin_shufflevector(lo, hi, 0, 1, 2, 3, 4, 5, 6, 7, 8, 9, 10, 11, 12, 13, 14, 15);
      }
#pragma unroll
      for (int tm = 0; tm < MT; ++tm)
#pragma unroll
        for (int tn = 0; tn < NT2; ++tn)
          acc[tm][tn] = __builtin_amdgcn_wmma_f32_16x16x32_bf16(
              false, af[tm], false, bfr[tn], (short)0, acc[tm][tn], false, false);
    }
  }
#endif

#pragma unroll
  for (int tm = 0; tm < MT; ++tm)
#pragma unroll
    for (int tn = 0; tn < NT2; ++tn)
#pragma unroll
      for (int i = 0; i < 8; ++i)
        Epi::op(g.ep, z, bh,
                m0 + wm * WM + tm * 16 + mOff + i,
                n0 + wn * WN + tn * 16 + lrow,
                acc[tm][tn][i]);
}

// ---------------- helper kernels ----------------
__global__ void f32_to_bf16_kernel(const float* __restrict__ in, __bf16* __restrict__ out, long n) {
  long i = ((long)blockIdx.x * blockDim.x + threadIdx.x) * 4;
  if (i < n) {
    float4 v = *(const float4*)(in + i);
    out[i + 0] = tob(v.x); out[i + 1] = tob(v.y);
    out[i + 2] = tob(v.z); out[i + 3] = tob(v.w);
  }
}

// tiled transpose fp32 (dim x dim) -> bf16; blockIdx.z selects source matrix
__global__ void transpose_to_bf16_kernel(const float* A0, const float* A1, const float* A2,
                                         __bf16* out, int dim) {
  const float* src = (blockIdx.z == 0) ? A0 : ((blockIdx.z == 1) ? A1 : A2);
  __bf16* dst = out + (long)blockIdx.z * dim * dim;
  __shared__ float tile[32][33];
  int x = blockIdx.x * 32 + threadIdx.x;
  int y = blockIdx.y * 32 + threadIdx.y;
#pragma unroll
  for (int i = 0; i < 32; i += 8) tile[threadIdx.y + i][threadIdx.x] = src[(long)(y + i) * dim + x];
  __syncthreads();
  x = blockIdx.y * 32 + threadIdx.x;
  y = blockIdx.x * 32 + threadIdx.y;
#pragma unroll
  for (int i = 0; i < 32; i += 8) dst[(long)(y + i) * dim + x] = tob(tile[threadIdx.x][threadIdx.y + i]);
}

__global__ void proj_pad_kernel(const float* __restrict__ proj, __bf16* __restrict__ out) {
  int i = blockIdx.x * blockDim.x + threadIdx.x;  // RP*HC
  if (i < RP * HC) {
    int r = i / HC;
    out[i] = (r < RREAL) ? tob(proj[i]) : tob(0.0f);
  }
}

__global__ void kvreduce_kernel(const float* __restrict__ part, __bf16* __restrict__ kvbf) {
  long idx = (long)blockIdx.x * blockDim.x + threadIdx.x;  // BH*HC*RP
  if (idx >= (long)BH * HC * RP) return;
  long bh = idx / (HC * RP);
  long off = idx - bh * (HC * RP);
  float s = 0.f;
#pragma unroll
  for (int c = 0; c < KCHUNKS; ++c)
    s += part[(bh * KCHUNKS + c) * (long)(HC * RP) + off];
  kvbf[idx] = tob(s);
}

__global__ void ksum_kernel(const __bf16* __restrict__ kpT, float* __restrict__ ksum) {
  long row = blockIdx.x;  // BH*RP rows, each NSEQ long (contiguous in kpT)
  const __bf16* p = kpT + row * NSEQ;
  float s = 0.f;
  for (int i = threadIdx.x; i < NSEQ; i += 128) s += (float)p[i];
  __shared__ float red[128];
  red[threadIdx.x] = s;
  __syncthreads();
  for (int o = 64; o > 0; o >>= 1) {
    if (threadIdx.x < o) red[threadIdx.x] += red[threadIdx.x + o];
    __syncthreads();
  }
  if (threadIdx.x == 0) ksum[row] = red[0];
}

__global__ void dinv_kernel(const __bf16* __restrict__ qp, const float* __restrict__ ksum,
                            float* __restrict__ dinv) {
  int wave = threadIdx.x >> 5, lane = threadIdx.x & 31;
  long tok = (long)blockIdx.x * 8 + wave;  // BH*NSEQ tokens
  if (tok >= (long)BH * NSEQ) return;
  long bh = tok >> 12;
  const __bf16* q = qp + tok * RP;
  const float* ks = ksum + bh * RP;
  float s = 0.f;
  for (int r = lane; r < RP; r += 32) s += (float)q[r] * ks[r];
#pragma unroll
  for (int off = 16; off > 0; off >>= 1) s += __shfl_down(s, off, 32);
  if (lane == 0) dinv[tok] = 1.0f / s;
}

// ---------------- launch ----------------
extern "C" void kernel_launch(void* const* d_in, const int* in_sizes, int n_in,
                              void* d_out, int out_size, void* d_ws, size_t ws_size,
                              hipStream_t stream) {
  const float* x    = (const float*)d_in[0];
  const float* Wq   = (const float*)d_in[1];
  const float* Wk   = (const float*)d_in[2];
  const float* Wv   = (const float*)d_in[3];
  const float* Wo   = (const float*)d_in[4];
  const float* bo   = (const float*)d_in[5];
  const float* proj = (const float*)d_in[6];
  float* out = (float*)d_out;

  char* w = (char*)d_ws;
  auto alloc = [&](size_t bytes) { void* p = (void*)w; w += (bytes + 255) & ~(size_t)255; return p; };

  __bf16* xb     = (__bf16*)alloc((size_t)TOK * CDIM * 2);
  __bf16* wqkvT  = (__bf16*)alloc((size_t)3072 * 1024 * 2);
  __bf16* woT    = (__bf16*)alloc((size_t)1024 * 1024 * 2);
  __bf16* projp  = (__bf16*)alloc((size_t)RP * HC * 2);
  __bf16* qk     = (__bf16*)alloc((size_t)TOK * 2048 * 2);
  __bf16* vT     = (__bf16*)alloc((size_t)BH * HC * NSEQ * 2);
  __bf16* qp     = (__bf16*)alloc((size_t)BH * NSEQ * RP * 2);
  __bf16* kpT    = (__bf16*)alloc((size_t)BH * RP * NSEQ * 2);
  float*  kvpart = (float*)alloc((size_t)BH * KCHUNKS * HC * RP * 4);
  __bf16* kvbf   = (__bf16*)alloc((size_t)BH * HC * RP * 2);
  float*  ksum   = (float*)alloc((size_t)BH * RP * 4);
  float*  dinv   = (float*)alloc((size_t)BH * NSEQ * 4);
  __bf16* ctx    = (__bf16*)alloc((size_t)TOK * CDIM * 2);

  // 1) conversions
  f32_to_bf16_kernel<<<(TOK * (long)CDIM / 4 + 255) / 256, 256, 0, stream>>>(x, xb, (long)TOK * CDIM);
  transpose_to_bf16_kernel<<<dim3(32, 32, 3), dim3(32, 8), 0, stream>>>(Wq, Wk, Wv, wqkvT, 1024);
  transpose_to_bf16_kernel<<<dim3(32, 32, 1), dim3(32, 8), 0, stream>>>(Wo, Wo, Wo, woT, 1024);
  proj_pad_kernel<<<(RP * HC + 255) / 256, 256, 0, stream>>>(proj, projp);

  // 2) QKV projection: [16384x1024] x [1024x3072]
  {
    GemmArgs g{};
    g.A = xb; g.lda = CDIM; g.sAb = 0; g.sAh = 0;
    g.Bt = wqkvT; g.ldb = CDIM; g.sBb = 0; g.sBh = 0;
    g.K = CDIM; g.kChunks = 1;
    g.ep.out0 = qk; g.ep.out1 = vT; g.ep.fin0 = nullptr;
    gemm_bf16_wmma<128, 128, 64, 32, EpiQKV><<<dim3(3072 / 128, TOK / 128, 1), 256, 0, stream>>>(g);
  }
  // 3) qp = relu(Q_head @ proj^T)+eps  (per bh)
  {
    GemmArgs g{};
    g.A = qk; g.lda = 2048; g.sAb = (long)NSEQ * 2048; g.sAh = 64;
    g.Bt = projp; g.ldb = HC; g.sBb = 0; g.sBh = 0;
    g.K = HC; g.kChunks = 1;
    g.ep.out0 = qp;
    gemm_bf16_wmma<128, 64, 64, 32, EpiQP><<<dim3(RP / 64, NSEQ / 128, BH), 128, 0, stream>>>(g);
  }
  // 4) kp^T (transposed store) (per bh)
  {
    GemmArgs g{};
    g.A = qk + 1024; g.lda = 2048; g.sAb = (long)NSEQ * 2048; g.sAh = 64;
    g.Bt = projp; g.ldb = HC; g.sBb = 0; g.sBh = 0;
    g.K = HC; g.kChunks = 1;
    g.ep.out0 = kpT;
    gemm_bf16_wmma<128, 64, 64, 32, EpiKP><<<dim3(RP / 64, NSEQ / 128, BH), 128, 0, stream>>>(g);
  }
  // 5) kv^T partials: vT[64x4096] x kpT[320x4096]^T, split-K over 8 chunks
  {
    GemmArgs g{};
    g.A = vT; g.lda = NSEQ; g.sAb = 16L * HC * NSEQ; g.sAh = (long)HC * NSEQ;
    g.Bt = kpT; g.ldb = NSEQ; g.sBb = 16L * RP * NSEQ; g.sBh = (long)RP * NSEQ;
    g.K = KCLEN; g.kChunks = KCHUNKS;
    g.ep.out0 = kvpart;
    gemm_bf16_wmma<64, 64, 32, 32, EpiKV><<<dim3(RP / 64, 1, BH * KCHUNKS), 128, 0, stream>>>(g);
  }
  kvreduce_kernel<<<((long)BH * HC * RP + 255) / 256, 256, 0, stream>>>(kvpart, kvbf);
  ksum_kernel<<<BH * RP, 128, 0, stream>>>(kpT, ksum);
  dinv_kernel<<<(BH * NSEQ) / 8, 256, 0, stream>>>(qp, ksum, dinv);

  // 6) qkv = qp @ kv, scaled by dinv, merged heads -> ctx
  {
    GemmArgs g{};
    g.A = qp; g.lda = RP; g.sAb = 16L * NSEQ * RP; g.sAh = (long)NSEQ * RP;
    g.Bt = kvbf; g.ldb = RP; g.sBb = 16L * HC * RP; g.sBh = (long)HC * RP;
    g.K = RP; g.kChunks = 1;
    g.ep.out0 = ctx; g.ep.fin0 = dinv;
    gemm_bf16_wmma<128, 64, 64, 32, EpiCtx><<<dim3(1, NSEQ / 128, BH), 128, 0, stream>>>(g);
  }
  // 7) out = ctx @ Wo + bo  (fp32 output)
  {
    GemmArgs g{};
    g.A = ctx; g.lda = CDIM; g.sAb = 0; g.sAh = 0;
    g.Bt = woT; g.ldb = CDIM; g.sBb = 0; g.sBh = 0;
    g.K = CDIM; g.kChunks = 1;
    g.ep.out0 = out; g.ep.fin0 = bo;
    gemm_bf16_wmma<128, 128, 64, 32, EpiOut><<<dim3(CDIM / 128, TOK / 128, 1), 256, 0, stream>>>(g);
  }
  (void)in_sizes; (void)n_in; (void)out_size; (void)ws_size;
}
